// PointerBlock_22050362097683
// MI455X (gfx1250) — matrix operand order repaired and checked by reading, not compiled, not run on hardware
//
#include <hip/hip_runtime.h>
#include <hip/hip_bf16.h>

typedef __attribute__((ext_vector_type(16))) _Float16 v16h;
typedef __attribute__((ext_vector_type(8)))  float    v8f;

#define LDS_STRIDE 40   // halves per row (32 data + 8 pad), 80B = 16B aligned

__device__ __forceinline__ float gelu_exact(float x) {
    return 0.5f * x * (1.0f + erff(x * 0.70710678118654752440f));
}

// ---------------------------------------------------------------------------
// Elementwise f32 -> f16
__global__ __launch_bounds__(256) void f32_to_f16_kernel(const float* __restrict__ in,
                                                         _Float16* __restrict__ out, int n) {
    int i = blockIdx.x * 256 + threadIdx.x;
    if (i < n) out[i] = (_Float16)in[i];
}

// Transposed pack: in [1024][1024] f32 row-major -> out[n][k] f16 (out = in^T)
__global__ __launch_bounds__(256) void pack_t_1024_kernel(const float* __restrict__ in,
                                                          _Float16* __restrict__ out) {
    int idx = blockIdx.x * 256 + threadIdx.x;          // 0 .. 1048575
    int n = idx >> 10;
    int k = idx & 1023;
    out[idx] = (_Float16)in[k * 1024 + n];
}

// vproj_w [H=16][D=1024][HD=64] f32 -> B^T [1024][1024] f16: out[(hi*64+k)*1024 + d]
__global__ __launch_bounds__(256) void pack_vproj_t_kernel(const float* __restrict__ in,
                                                           _Float16* __restrict__ out) {
    int idx = blockIdx.x * 256 + threadIdx.x;          // 0 .. 1048575
    int n  = idx >> 10;            // output column of original B = hi*64+k
    int d  = idx & 1023;
    int hi = n >> 6;
    int k  = n & 63;
    out[idx] = (_Float16)in[hi * 65536 + d * 64 + k];
}

// fus_w1 [H][192][64] rows 0..63 -> transposed per head: out[hi][j][r] f16
__global__ __launch_bounds__(256) void pack_w1a_t_kernel(const float* __restrict__ in,
                                                         _Float16* __restrict__ out) {
    int idx = blockIdx.x * 256 + threadIdx.x;          // 0 .. 65535
    int hi = idx >> 12;
    int j  = (idx >> 6) & 63;
    int r  = idx & 63;
    out[idx] = (_Float16)in[hi * 12288 + r * 64 + j];
}

// fus_w2 [H][64][64] -> transposed per head: out[hi][j][r] f16
__global__ __launch_bounds__(256) void pack_w2_t_kernel(const float* __restrict__ in,
                                                        _Float16* __restrict__ out) {
    int idx = blockIdx.x * 256 + threadIdx.x;          // 0 .. 65535
    int hi = idx >> 12;
    int j  = (idx >> 6) & 63;
    int r  = idx & 63;
    out[idx] = (_Float16)in[hi * 4096 + r * 64 + j];
}

// Column mean over N=2048 tokens: hf32 [B=2][2048][1024] -> mhf [2][1024]
__global__ __launch_bounds__(256) void colmean_kernel(const float* __restrict__ hf,
                                                      float* __restrict__ mhf) {
    __shared__ float red[256];
    int col = blockIdx.x;                 // 0..2047
    int b = col >> 10, c = col & 1023;
    const float* p = hf + (size_t)b * 2048 * 1024 + c;
    float s = 0.f;
    for (int n = threadIdx.x; n < 2048; n += 256) s += p[(size_t)n * 1024];
    red[threadIdx.x] = s;
    __syncthreads();
    for (int st = 128; st > 0; st >>= 1) {
        if (threadIdx.x < st) red[threadIdx.x] += red[threadIdx.x + st];
        __syncthreads();
    }
    if (threadIdx.x == 0) mhf[col] = red[0] * (1.0f / 2048.0f);
}

// cvec[b][hi][j] = fus_b1[hi][j] + sum_k mhf[b][hi*64+k] * (W1[hi][64+k][j] + W1[hi][128+k][j])
__global__ __launch_bounds__(256) void cvec_kernel(const float* __restrict__ mhf,
                                                   const float* __restrict__ w1,
                                                   const float* __restrict__ b1,
                                                   float* __restrict__ cvec) {
    int idx = blockIdx.x * 256 + threadIdx.x;     // 0..2047
    if (idx >= 2048) return;
    int b = idx >> 10, hi = (idx >> 6) & 15, j = idx & 63;
    const float* m  = mhf + b * 1024 + hi * 64;
    const float* wB = w1 + hi * 12288 + 64 * 64 + j;
    const float* wC = w1 + hi * 12288 + 128 * 64 + j;
    float s = b1[hi * 64 + j];
    for (int k = 0; k < 64; ++k) s += m[k] * (wB[k * 64] + wC[k * 64]);
    cvec[idx] = s;
}

// Constant outputs: main_fwd_ptr = 1023 (as float), avg_strength = 1 - N*(1/N)*log(1/N+eps)
__global__ __launch_bounds__(256) void tail_kernel(float* __restrict__ dout) {
    int i = blockIdx.x * 256 + threadIdx.x;
    if (i >= 4096) return;
    const float pf = 1.0f / 2048.0f;
    float t = pf * logf(pf + 1e-8f);
    dout[4194304 + i] = 1023.0f;
    dout[4198400 + i] = 1.0f - 2048.0f * t;
}

// ---------------------------------------------------------------------------
// WMMA GEMM: C = A @ B (+bias)(gelu?), f16 in / f32 acc.
// B is supplied TRANSPOSED (Bt[n][k] row-major), so both operands stage into
// LDS with b128 traffic only. Block = 128 threads (4 wave32); block tile
// 128(M) x 64(N); each wave computes 32x64 = 8 WMMA tiles per K-step.
// z batches per-head slices (a_colblk / b_zstride / c_colblk).
__global__ __launch_bounds__(128)
void wmma_gemm_kernel(const _Float16* __restrict__ A,
                      const _Float16* __restrict__ Bt,
                      const float* __restrict__ bias, int bias_bstride,
                      float* __restrict__ Cf, _Float16* __restrict__ Ch,
                      int K, int lda, int ldbt, int ldc,
                      int a_colblk, int b_zstride, int c_colblk, int flags) {
    __shared__ __align__(16) _Float16 lsA[128 * LDS_STRIDE];   // 10240 B
    __shared__ __align__(16) _Float16 lsB[64 * LDS_STRIDE];    //  5120 B

    const int tid  = threadIdx.x;
    const int wave = tid >> 5;
    const int lane = tid & 31;
    const int lg   = lane & 15;
    const int hl   = lane >> 4;           // 0: lanes 0-15, 1: lanes 16-31

    const int m0 = blockIdx.y * 128;
    const int z  = blockIdx.z;
    const int n0 = blockIdx.x * 64;
    const int a_col0 = z * a_colblk;
    const _Float16* Bz = Bt + (size_t)z * b_zstride;
    const int c0 = z * c_colblk + n0;

    // staging indices: A -> 1 row (32 halves) per thread; Bt -> half row per thread
    const int brow = tid >> 1;            // 0..63
    const int bseg = (tid & 1) << 4;      // 0 or 16

    union Frag { uint4 u[2]; v16h v; };
    v8f acc[8] = {};                      // [mi][nt] -> acc[mi*4+nt]

    for (int k0 = 0; k0 < K; k0 += 32) {
        // ---- stage A tile (128 rows x 32 k), row-major
        {
            const uint4* src = (const uint4*)(A + (size_t)(m0 + tid) * lda + a_col0 + k0);
            uint4 a0 = src[0], a1 = src[1], a2 = src[2], a3 = src[3];
            uint4* dst = (uint4*)(&lsA[tid * LDS_STRIDE]);
            dst[0] = a0; dst[1] = a1; dst[2] = a2; dst[3] = a3;
            if (k0 + 32 < K)
                __builtin_prefetch(A + (size_t)(m0 + tid) * lda + a_col0 + k0 + 32, 0, 3);
        }
        // ---- stage Bt tile (64 n-rows x 32 k), row-major (already transposed)
        {
            const uint4* src = (const uint4*)(Bz + (size_t)(n0 + brow) * ldbt + k0 + bseg);
            uint4 b0 = src[0], b1 = src[1];
            uint4* dst = (uint4*)(&lsB[brow * LDS_STRIDE + bseg]);
            dst[0] = b0; dst[1] = b1;
            if (k0 + 32 < K)
                __builtin_prefetch(Bz + (size_t)(n0 + brow) * ldbt + k0 + 32 + bseg, 0, 3);
        }
        __syncthreads();

        // ---- load ALL fragments first (distinct regs -> WMMAs can pipeline)
        Frag af[2], bf[4];
#pragma unroll
        for (int mi = 0; mi < 2; ++mi) {
            const _Float16* ar = &lsA[(wave * 32 + mi * 16 + lg) * LDS_STRIDE];
            af[mi].u[0] = *(const uint4*)(ar + hl * 8);
            af[mi].u[1] = *(const uint4*)(ar + 16 + hl * 8);
        }
#pragma unroll
        for (int nt = 0; nt < 4; ++nt) {
            const _Float16* bc = &lsB[(nt * 16 + lg) * LDS_STRIDE];
            bf[nt].u[0] = *(const uint4*)(bc + hl * 8);
            bf[nt].u[1] = *(const uint4*)(bc + 16 + hl * 8);
        }
        // ---- 8 back-to-back WMMAs
#pragma unroll
        for (int mi = 0; mi < 2; ++mi)
#pragma unroll
            for (int nt = 0; nt < 4; ++nt)
                acc[mi * 4 + nt] = __builtin_amdgcn_wmma_f32_16x16x32_f16(
                    false, af[mi].v, false, bf[nt].v, (short)0, acc[mi * 4 + nt],
                    false, false);
        __syncthreads();
    }

    // ---- epilogue: VGPR i of v8f -> lanes0-15 row=i, lanes16-31 row=i+8; col = lane&15
#pragma unroll
    for (int mi = 0; mi < 2; ++mi) {
        const int row_base = m0 + wave * 32 + mi * 16 + hl * 8;
#pragma unroll
        for (int nt = 0; nt < 4; ++nt) {
            const int jcol = nt * 16 + lg;        // column within the z-slice
            const int col  = c0 + jcol;
#pragma unroll
            for (int i = 0; i < 8; ++i) {
                const int r = row_base + i;
                float v = acc[mi * 4 + nt][i];
                if (bias) v += bias[(r >> 11) * bias_bstride + z * 64 + jcol];
                if (flags & 1) v = gelu_exact(v);
                const size_t off = (size_t)r * ldc + col;
                if (Cf) Cf[off] = v;
                if (Ch) Ch[off] = (_Float16)v;
            }
        }
    }
}

// ---------------------------------------------------------------------------
extern "C" void kernel_launch(void* const* d_in, const int* in_sizes, int n_in,
                              void* d_out, int out_size, void* d_ws, size_t ws_size,
                              hipStream_t stream) {
    // setup_inputs order. fe_*/be_* (indices 1..8) are DEAD CODE (uniform softmax).
    const float* h        = (const float*)d_in[0];
    const float* vproj_w  = (const float*)d_in[9];
    const float* fus_w1   = (const float*)d_in[10];
    const float* fus_b1   = (const float*)d_in[11];
    const float* fus_w2   = (const float*)d_in[12];
    const float* fus_b2   = (const float*)d_in[13];
    const float* out_proj = (const float*)d_in[14];
    const float* o_proj   = (const float*)d_in[15];
    float* out = (float*)d_out;

    char* ws = (char*)d_ws;
    size_t off = 0;
    auto alloc = [&](size_t bytes) -> char* {
        char* p = ws + off;
        off += (bytes + 255) & ~(size_t)255;
        return p;
    };
    _Float16* h16      = (_Float16*)alloc(4194304ull * 2);   // [4096][1024]
    _Float16* vprojT16 = (_Float16*)alloc(1048576ull * 2);   // [1024][1024] (B^T)
    _Float16* outpT16  = (_Float16*)alloc(1048576ull * 2);
    _Float16* oprojT16 = (_Float16*)alloc(1048576ull * 2);
    _Float16* w1aT16   = (_Float16*)alloc(65536ull * 2);     // [16][64][64] (per-head B^T)
    _Float16* w2T16    = (_Float16*)alloc(65536ull * 2);
    float*    mhf      = (float*)alloc(2048ull * 4);         // [2][1024]
    float*    cvec     = (float*)alloc(2048ull * 4);         // [2][16][64]
    float*    hf32     = (float*)alloc(4194304ull * 4);      // [4096][1024]
    _Float16* hf16     = (_Float16*)alloc(4194304ull * 2);
    _Float16* t16      = (_Float16*)alloc(4194304ull * 2);
    _Float16* multi16  = (_Float16*)alloc(4194304ull * 2);
    _Float16* out16    = (_Float16*)alloc(4194304ull * 2);

    // 1) pack weights / activations to f16 (weights transposed for b128-only staging)
    f32_to_f16_kernel<<<16384, 256, 0, stream>>>(h, h16, 4194304);
    pack_vproj_t_kernel<<<4096, 256, 0, stream>>>(vproj_w, vprojT16);
    pack_t_1024_kernel<<<4096, 256, 0, stream>>>(out_proj, outpT16);
    pack_t_1024_kernel<<<4096, 256, 0, stream>>>(o_proj, oprojT16);
    pack_w1a_t_kernel<<<256, 256, 0, stream>>>(fus_w1, w1aT16);
    pack_w2_t_kernel<<<256, 256, 0, stream>>>(fus_w2, w2T16);

    // 2) hf = h @ vproj  (dense 4096x1024x1024), keep f32 (for mean) + f16 (next A)
    wmma_gemm_kernel<<<dim3(16, 32, 1), 128, 0, stream>>>(
        h16, vprojT16, nullptr, 0, hf32, hf16,
        1024, 1024, 1024, 1024, 0, 0, 0, 0);

    // 3) per-batch column means, folded into per-(batch,head) constant bias cvec
    colmean_kernel<<<2048, 256, 0, stream>>>(hf32, mhf);
    cvec_kernel<<<8, 256, 0, stream>>>(mhf, fus_w1, fus_b1, cvec);

    // 4) fusion layer1 per head: t = gelu(hf_head @ W1a + cvec)   (z = head, K = 64)
    wmma_gemm_kernel<<<dim3(1, 32, 16), 128, 0, stream>>>(
        hf16, w1aT16, cvec, 1024, nullptr, t16,
        64, 1024, 64, 1024, 64, 4096, 64, 1);

    // 5) fusion layer2 per head: multi = t_head @ W2 + b2
    wmma_gemm_kernel<<<dim3(1, 32, 16), 128, 0, stream>>>(
        t16, w2T16, fus_b2, 0, nullptr, multi16,
        64, 1024, 64, 1024, 64, 4096, 64, 0);

    // 6) out = multi @ out_proj
    wmma_gemm_kernel<<<dim3(16, 32, 1), 128, 0, stream>>>(
        multi16, outpT16, nullptr, 0, nullptr, out16,
        1024, 1024, 1024, 1024, 0, 0, 0, 0);

    // 7) z = out @ o_proj -> f32 into d_out[0 : 4194304]
    wmma_gemm_kernel<<<dim3(16, 32, 1), 128, 0, stream>>>(
        out16, oprojT16, nullptr, 0, out, nullptr,
        1024, 1024, 1024, 1024, 0, 0, 0, 0);

    // 8) constant pointer / strength outputs
    tail_kernel<<<16, 256, 0, stream>>>(out);

    (void)in_sizes; (void)n_in; (void)out_size; (void)ws_size;
}
